// PositionAttentionModule_72653666779527
// MI455X (gfx1250) — matrix operand order repaired and checked by reading, not compiled, not run on hardware
//
#include <hip/hip_runtime.h>
#include <math.h>

// ---------------------------------------------------------------------------
// PositionAttentionModule fused WMMA implementation for gfx1250 (MI455X)
//   B=4, C=512, MID=64, H=W=64 -> N=4096
//   V tile streamed into LDS by the Tensor Data Mover (TDM), double-buffered
//   and overlapped with WMMA compute (s_wait_tensorcnt).
// ---------------------------------------------------------------------------

typedef __attribute__((ext_vector_type(16))) __bf16 v16bf;
typedef __attribute__((ext_vector_type(8)))  float  v8f;
typedef __attribute__((ext_vector_type(4)))  unsigned int v4u;
typedef __attribute__((ext_vector_type(8)))  int    v8i;
typedef __attribute__((ext_vector_type(4)))  int    v4i;
typedef unsigned short u16;
typedef unsigned int   u32;
typedef unsigned long long u64;

#define CIN   512
#define MIDW  64
#define BATCH 4
#define NPIX  4096

// float -> bf16 (round-to-nearest-even), stored as raw u16
__device__ __forceinline__ u16 f2bf(float f) {
  u32 u = __float_as_uint(f);
  u32 r = (u + 0x7FFFu + ((u >> 16) & 1u)) >> 16;
  return (u16)r;
}

union FragAB { uint4 q[2]; v16bf v; };

__device__ __forceinline__ v8f zero8() {
  v8f z = {0.f, 0.f, 0.f, 0.f, 0.f, 0.f, 0.f, 0.f};
  return z;
}

__device__ __forceinline__ v8f wmma_bf16(v16bf a, v16bf b, v8f c) {
  // D = A(16x32 bf16) * B(32x16 bf16) + C(16x16 f32)
  return __builtin_amdgcn_wmma_f32_16x16x32_bf16(
      /*neg_a=*/false, a, /*neg_b=*/false, b,
      /*c_mod=*/(short)0, c, /*reuse_a=*/false, /*reuse_b=*/false);
}

// A fragment, 16x32 bf16, row-major source (row stride ld elements).
// lane L: row = row0 + L%16; elements 0..7 = k0 + (L/16)*8 .. +7,
//         elements 8..15 = k0 + 16 + (L/16)*8 .. +7   (per ISA table)
__device__ __forceinline__ v16bf load_A16x32(const u16* p, int ld, int row0, int k0) {
  int lane = threadIdx.x & 31;
  const u16* base = p + (size_t)(row0 + (lane & 15)) * ld + k0 + ((lane >> 4) << 3);
  FragAB f;
  f.q[0] = *(const uint4*)(base);
  f.q[1] = *(const uint4*)(base + 16);
  return f.v;
}

// B fragment, 32x16 bf16, source stored "n-major": element (k,n) at p[n*ld + k].
// lane L: n = n0 + L%16; elements 0..7 = k0 + (L/16)*8 .. +7,
//         elements 8..15 = k0 + 16 + (L/16)*8 .. +7   (per ISA table)
__device__ __forceinline__ v16bf load_B32x16(const u16* p, int ld, int n0, int k0) {
  int lane = threadIdx.x & 31;
  const u16* base = p + (size_t)(n0 + (lane & 15)) * ld + k0 + ((lane >> 4) << 3);
  FragAB f;
  f.q[0] = *(const uint4*)(base);
  f.q[1] = *(const uint4*)(base + 16);
  return f.v;
}

// Generic LDS pointer -> byte offset within LDS (flat LDS aperture truncation)
__device__ __forceinline__ u32 lds_off(const void* p) {
  return (u32)(uintptr_t)p;
}

// ---------------------------------------------------------------------------
// TDM: async 2D tile load (global -> LDS). Builds D# groups 0/1 per ISA 8.3/8.4:
//  g0: [1:0]=count=1, [63:32]=lds_addr, [120:64]=global_addr, [127:126]=type 2
//  g1: [17:16]=data_size (1 => 2 bytes), tensor_dim0/1, tile_dim0/1, stride0
// Tracked on TENSORcnt. Issued uniformly by one wave.
// (6-arg builtin form on this toolchain: g0, g1, g2, g3, g4, cpol)
// ---------------------------------------------------------------------------
__device__ __forceinline__ void tdm_load_2d_bf16(u32 ldsDst, const u16* gsrc,
                                                 u32 tensorD0, u32 tensorD1,
                                                 u32 tileD0, u32 tileD1,
                                                 u64 stride0) {
  u64 ga = (u64)(uintptr_t)gsrc;
  v4u g0;
  g0.x = 1u;                                        // count=1, user desc
  g0.y = ldsDst;                                    // lds_addr (bytes)
  g0.z = (u32)ga;                                   // global_addr[31:0]
  g0.w = (u32)((ga >> 32) & 0x01FFFFFFu) | (2u << 30); // addr[56:32] | type=2
  u32 w0 = (1u << 16);                              // data_size = 2 bytes
  u32 w1 = (tensorD0 & 0xFFFFu) << 16;              // tensor_dim0[15:0]
  u32 w2 = ((tensorD0 >> 16) & 0xFFFFu) | ((tensorD1 & 0xFFFFu) << 16);
  u32 w3 = ((tensorD1 >> 16) & 0xFFFFu) | ((tileD0 & 0xFFFFu) << 16);
  u32 w4 = (tileD1 & 0xFFFFu);                      // tile_dim1 (tile_dim2=0)
  u32 w5 = (u32)(stride0 & 0xFFFFFFFFu);            // tensor_dim0_stride[31:0]
  u32 w6 = (u32)((stride0 >> 32) & 0xFFFFu);        // stride[47:32], dim1_stride=0
  u32 w7 = 0u;
  v8i g1 = {(int)w0, (int)w1, (int)w2, (int)w3, (int)w4, (int)w5, (int)w6, (int)w7};
  v4i gz4 = {0, 0, 0, 0};
  v8i gz8 = {0, 0, 0, 0, 0, 0, 0, 0};
  __builtin_amdgcn_tensor_load_to_lds(g0, g1, gz4, gz4, gz8, 0);
}

// ---------------------------------------------------------------------------
// Kernel 0a: f32 -> bf16 weight conversion
// ---------------------------------------------------------------------------
__global__ void k_convert(const float* __restrict__ s, u16* __restrict__ d, int n) {
  int i = blockIdx.x * blockDim.x + threadIdx.x;
  if (i < n) d[i] = f2bf(s[i]);
}

// ---------------------------------------------------------------------------
// Kernel 0b: x[b][c][n] f32 -> xt[b][n][c] bf16 (LDS-tiled transpose)
// ---------------------------------------------------------------------------
__global__ void k_transpose(const float* __restrict__ x, u16* __restrict__ xt) {
  __shared__ float t[32][33];
  int b  = blockIdx.z;
  int n0 = blockIdx.x * 32;
  int c0 = blockIdx.y * 32;
  const float* xb = x + (size_t)b * CIN * NPIX;
  #pragma unroll
  for (int i = threadIdx.y; i < 32; i += 8)
    t[i][threadIdx.x] = xb[(size_t)(c0 + i) * NPIX + n0 + threadIdx.x];
  __syncthreads();
  u16* xtb = xt + (size_t)b * NPIX * CIN;
  #pragma unroll
  for (int i = threadIdx.y; i < 32; i += 8)
    xtb[(size_t)(n0 + i) * CIN + c0 + threadIdx.x] = f2bf(t[threadIdx.x][i]);
}

// ---------------------------------------------------------------------------
// Kernel 1: Q/K projections.  Qt[b][n][m] = sum_c xt[b][n][c]*Wq[m][c] + bq[m]
// One wave per 16 rows of n; m covers all 64 (4 tiles); K-loop over 512.
// ---------------------------------------------------------------------------
__global__ __launch_bounds__(256)
void k_proj_qk(const u16* __restrict__ xt,
               const u16* __restrict__ wq, const u16* __restrict__ wk,
               const float* __restrict__ bq, const float* __restrict__ bk,
               u16* __restrict__ Qt, u16* __restrict__ Kt) {
  int wid = blockIdx.x * (blockDim.x >> 5) + (threadIdx.x >> 5);
  const int rowsPerB = NPIX / 16;  // 256
  int b  = wid / rowsPerB;
  int n0 = (wid % rowsPerB) * 16;
  const u16* xtb = xt + (size_t)b * NPIX * CIN;

  v8f dq[4], dk[4];
  #pragma unroll
  for (int t = 0; t < 4; ++t) { dq[t] = zero8(); dk[t] = zero8(); }

  for (int kc = 0; kc < CIN; kc += 32) {
    v16bf a = load_A16x32(xtb, CIN, n0, kc);
    #pragma unroll
    for (int mt = 0; mt < 4; ++mt) {
      v16bf bqf = load_B32x16(wq, CIN, mt * 16, kc);
      dq[mt] = wmma_bf16(a, bqf, dq[mt]);
      v16bf bkf = load_B32x16(wk, CIN, mt * 16, kc);
      dk[mt] = wmma_bf16(a, bkf, dk[mt]);
    }
  }

  int lane = threadIdx.x & 31;
  int nn   = (lane & 15);
  int mb   = (lane >> 4) << 3;  // +8 rows for upper half-lanes
  u16* Qtb = Qt + (size_t)b * NPIX * MIDW;
  u16* Ktb = Kt + (size_t)b * NPIX * MIDW;
  #pragma unroll
  for (int mt = 0; mt < 4; ++mt) {
    float biq = bq[mt * 16 + nn];
    float bik = bk[mt * 16 + nn];
    #pragma unroll
    for (int r = 0; r < 8; ++r) {
      int row = n0 + r + mb;           // D's M index = pixel n
      int col = mt * 16 + nn;          // D's N index = proj channel m
      Qtb[(size_t)row * MIDW + col] = f2bf(dq[mt][r] + biq);
      Ktb[(size_t)row * MIDW + col] = f2bf(dk[mt][r] + bik);
    }
  }
}

// ---------------------------------------------------------------------------
// Kernel 2: V projection.  V[b][c][n] = sum_c' Wv[c][c']*xt[b][n][c'] + bv[c]
// One wave: 64 channels (4 tiles) x 16 pixels.
// ---------------------------------------------------------------------------
__global__ __launch_bounds__(256)
void k_proj_v(const u16* __restrict__ xt, const u16* __restrict__ wv,
              const float* __restrict__ bv, u16* __restrict__ V) {
  int wid = blockIdx.x * (blockDim.x >> 5) + (threadIdx.x >> 5);
  const int nTiles    = NPIX / 16;             // 256
  const int tilesPerB = (CIN / 64) * nTiles;   // 2048
  int b  = wid / tilesPerB;
  int t  = wid % tilesPerB;
  int c0 = (t / nTiles) * 64;
  int n0 = (t % nTiles) * 16;
  const u16* xtb = xt + (size_t)b * NPIX * CIN;

  v8f acc[4];
  #pragma unroll
  for (int i = 0; i < 4; ++i) acc[i] = zero8();

  for (int kc = 0; kc < CIN; kc += 32) {
    v16bf bfrag = load_B32x16(xtb, CIN, n0, kc);   // B[k=c'][n] = xt[n][c']
    #pragma unroll
    for (int ct = 0; ct < 4; ++ct) {
      v16bf afrag = load_A16x32(wv, CIN, c0 + ct * 16, kc);
      acc[ct] = wmma_bf16(afrag, bfrag, acc[ct]);
    }
  }

  int lane = threadIdx.x & 31;
  int nn   = n0 + (lane & 15);
  int mb   = (lane >> 4) << 3;
  u16* Vb = V + (size_t)b * CIN * NPIX;
  #pragma unroll
  for (int ct = 0; ct < 4; ++ct)
    #pragma unroll
    for (int r = 0; r < 8; ++r) {
      int c = c0 + ct * 16 + r + mb;
      Vb[(size_t)c * NPIX + nn] = f2bf(acc[ct][r] + bv[c]);
    }
}

// ---------------------------------------------------------------------------
// Kernel 3: fused flash attention + output.
// Grid: (N/64, B). Block: 256 threads = 8 waves.
//   wave w: i-group = w>>1 (16 query rows), channel half = (w&1)*256.
// Per 64-wide j tile:
//   * TDM streams V[:,j0:j0+64] (512x64 bf16, 64KB) into a 2-deep LDS pipe,
//     issued by wave 0; s_wait_tensorcnt(1) overlaps DMA with compute.
//   * S = Q*K^T (WMMA, K fragments direct from global),
//   * online softmax (wave32 shfl reductions),
//   * P -> LDS bf16 -> reloaded as B fragments (transpose via LDS),
//   * O += V * P^T (WMMA, A fragments from the TDM-filled LDS tile).
// Epilogue: out = gamma*O/l + x.
// ---------------------------------------------------------------------------
__global__ __launch_bounds__(256)
void k_attn(const u16* __restrict__ Qt, const u16* __restrict__ Kt,
            const u16* __restrict__ V,  const float* __restrict__ x,
            const float* __restrict__ gamma, float* __restrict__ out) {
  __shared__ __align__(16) u16 sV[2][CIN][64];  // 2 x 64KB V tiles (TDM dest)
  __shared__ __align__(16) u16 sP[8][16][64];   // per-wave P tile (16 x 64 bf16)
  __shared__ float sB[8][16];                   // per-wave row broadcast

  int b    = blockIdx.y;
  int w    = threadIdx.x >> 5;
  int lane = threadIdx.x & 31;
  int half = lane >> 4;           // 0/1
  int ln   = lane & 15;
  int i0   = blockIdx.x * 64 + (w >> 1) * 16;
  int chalf = (w & 1) * 256;

  const u16* Qtb = Qt + (size_t)b * NPIX * MIDW;
  const u16* Ktb = Kt + (size_t)b * NPIX * MIDW;
  const u16* Vb  = V  + (size_t)b * CIN * NPIX;

  // Q fragments for this wave's 16 rows (K = 0..63 in two chunks), loop-invariant
  v16bf qa0 = load_A16x32(Qtb, MIDW, i0, 0);
  v16bf qa1 = load_A16x32(Qtb, MIDW, i0, 32);

  v8f acc[16];
  #pragma unroll
  for (int t = 0; t < 16; ++t) acc[t] = zero8();

  float mr[8], lr[8];
  #pragma unroll
  for (int r = 0; r < 8; ++r) { mr[r] = -INFINITY; lr[r] = 0.f; }

  const int NTILES = NPIX / 64;   // 64 j-tiles

  // Prologue: kick off DMA of the first V tile.
  if (w == 0)
    tdm_load_2d_bf16(lds_off(&sV[0][0][0]), Vb + 0,
                     /*tensorD0=*/NPIX, /*tensorD1=*/CIN,
                     /*tileD0=*/64, /*tileD1=*/CIN, /*stride0=*/NPIX);

  for (int jt = 0; jt < NTILES; ++jt) {
    int cur = jt & 1;
    int j0  = jt * 64;
    if (w == 0) {
      if (jt + 1 < NTILES) {
        tdm_load_2d_bf16(lds_off(&sV[cur ^ 1][0][0]), Vb + (size_t)(jt + 1) * 64,
                         NPIX, CIN, 64, CIN, NPIX);
        // wait for the current tile; only the just-issued DMA stays in flight
        __builtin_amdgcn_s_wait_tensorcnt(1);
      } else {
        __builtin_amdgcn_s_wait_tensorcnt(0);
      }
    }
    __syncthreads();   // current V tile visible to all waves

    // ---- scores: S[16 x 64] in four 16x16 fragments
    v8f s[4];
    #pragma unroll
    for (int jj = 0; jj < 4; ++jj) {
      v16bf b0 = load_B32x16(Ktb, MIDW, j0 + jj * 16, 0);
      v16bf b1 = load_B32x16(Ktb, MIDW, j0 + jj * 16, 32);
      v8f z = zero8();
      z = wmma_bf16(qa0, b0, z);
      z = wmma_bf16(qa1, b1, z);
      s[jj] = z;
    }

    // ---- online softmax. Fragment element (r,lane): row i = r + 8*half,
    //      col j = j0 + jj*16 + ln. Row reductions = xor-shuffles over 16 lanes.
    #pragma unroll
    for (int r = 0; r < 8; ++r) {
      float mx = fmaxf(fmaxf(s[0][r], s[1][r]), fmaxf(s[2][r], s[3][r]));
      mx = fmaxf(mx, __shfl_xor(mx, 1, 32));
      mx = fmaxf(mx, __shfl_xor(mx, 2, 32));
      mx = fmaxf(mx, __shfl_xor(mx, 4, 32));
      mx = fmaxf(mx, __shfl_xor(mx, 8, 32));
      float mnew  = fmaxf(mr[r], mx);
      float alpha = __expf(mr[r] - mnew);
      mr[r] = mnew;

      float rs = 0.f;
      #pragma unroll
      for (int jj = 0; jj < 4; ++jj) {
        float pv = __expf(s[jj][r] - mnew);
        rs += pv;
        sP[w][r + (half << 3)][jj * 16 + ln] = f2bf(pv);
      }
      rs += __shfl_xor(rs, 1, 32);
      rs += __shfl_xor(rs, 2, 32);
      rs += __shfl_xor(rs, 4, 32);
      rs += __shfl_xor(rs, 8, 32);
      lr[r] = lr[r] * alpha + rs;
      sB[w][r + (half << 3)] = alpha;   // broadcast alpha per row i
    }

    // rescale accumulators by alpha of row i = ln (LDS ops are in-order per wave)
    float av = sB[w][ln];
    #pragma unroll
    for (int t = 0; t < 16; ++t)
      #pragma unroll
      for (int r = 0; r < 8; ++r) acc[t][r] *= av;

    // P^T fragments out of LDS: B[k=j_local][n=i] = P[i][j_local]
    v16bf pb0 = load_B32x16(&sP[w][0][0], 64, 0, 0);
    v16bf pb1 = load_B32x16(&sP[w][0][0], 64, 0, 32);

    // ---- O += V * P^T over this wave's 256 channels, V from TDM-filled LDS
    const u16* sVc = &sV[cur][0][0];
    #pragma unroll
    for (int t = 0; t < 16; ++t) {
      int c0 = chalf + t * 16;
      v16bf a0 = load_A16x32(sVc, 64, c0, 0);
      v16bf a1 = load_A16x32(sVc, 64, c0, 32);
      acc[t] = wmma_bf16(a0, pb0, acc[t]);
      acc[t] = wmma_bf16(a1, pb1, acc[t]);
    }

    __syncthreads();   // all waves done with sV[cur] before it is re-filled
  }

  // ---- epilogue: divide by softmax sum, out = gamma*O + x
  #pragma unroll
  for (int r = 0; r < 8; ++r) sB[w][r + (half << 3)] = lr[r];
  float linv = 1.f / sB[w][ln];
  float gm   = gamma[0];

  int i = i0 + ln;
  const float* xb = x   + (size_t)b * CIN * NPIX;
  float*       ob = out + (size_t)b * CIN * NPIX;
  #pragma unroll
  for (int t = 0; t < 16; ++t)
    #pragma unroll
    for (int r = 0; r < 8; ++r) {
      int c = chalf + t * 16 + r + (half << 3);
      size_t idx = (size_t)c * NPIX + i;
      ob[idx] = gm * acc[t][r] * linv + xb[idx];
    }
}

// ---------------------------------------------------------------------------
// Host launcher
// ---------------------------------------------------------------------------
extern "C" void kernel_launch(void* const* d_in, const int* in_sizes, int n_in,
                              void* d_out, int out_size, void* d_ws, size_t ws_size,
                              hipStream_t stream) {
  const float* x     = (const float*)d_in[0];
  const float* Wq    = (const float*)d_in[1];
  const float* bq    = (const float*)d_in[2];
  const float* Wk    = (const float*)d_in[3];
  const float* bk    = (const float*)d_in[4];
  const float* Wv    = (const float*)d_in[5];
  const float* bv    = (const float*)d_in[6];
  const float* gamma = (const float*)d_in[7];
  float* out = (float*)d_out;

  char* ws = (char*)d_ws;
  size_t off = 0;
  auto carve = [&](size_t bytes) -> void* {
    void* p = ws + off;
    off += (bytes + 255) & ~(size_t)255;
    return p;
  };
  u16* xt  = (u16*)carve((size_t)BATCH * NPIX * CIN * 2);  // x transposed, bf16
  u16* wqb = (u16*)carve((size_t)MIDW * CIN * 2);
  u16* wkb = (u16*)carve((size_t)MIDW * CIN * 2);
  u16* wvb = (u16*)carve((size_t)CIN * CIN * 2);
  u16* Qt  = (u16*)carve((size_t)BATCH * NPIX * MIDW * 2);
  u16* Kt  = (u16*)carve((size_t)BATCH * NPIX * MIDW * 2);
  u16* Vp  = (u16*)carve((size_t)BATCH * CIN * NPIX * 2);
  (void)ws_size; (void)in_sizes; (void)n_in; (void)out_size;

  // 0a: weight conversion
  {
    int n1 = MIDW * CIN, n2 = CIN * CIN;
    k_convert<<<(n1 + 255) / 256, 256, 0, stream>>>(Wq, wqb, n1);
    k_convert<<<(n1 + 255) / 256, 256, 0, stream>>>(Wk, wkb, n1);
    k_convert<<<(n2 + 255) / 256, 256, 0, stream>>>(Wv, wvb, n2);
  }
  // 0b: x transpose + bf16
  k_transpose<<<dim3(NPIX / 32, CIN / 32, BATCH), dim3(32, 8), 0, stream>>>(x, xt);

  // 1: Q/K projections — B*N/16 = 1024 waves, 8 waves/block
  k_proj_qk<<<(BATCH * NPIX / 16) / 8, 256, 0, stream>>>(xt, wqb, wkb, bq, bk, Qt, Kt);

  // 2: V projection — B*(C/64)*(N/16) = 8192 waves, 8 waves/block
  k_proj_v<<<(BATCH * (CIN / 64) * (NPIX / 16)) / 8, 256, 0, stream>>>(xt, wvb, bv, Vp);

  // 3: fused attention + output (TDM-fed, double-buffered V tiles)
  k_attn<<<dim3(NPIX / 64, BATCH), 256, 0, stream>>>(Qt, Kt, Vp, x, gamma, out);
}
